// TEAugAdapterCLIPAttention_72206990180693
// MI455X (gfx1250) — matrix operand order repaired and checked by reading, not compiled, not run on hardware
//
#include <hip/hip_runtime.h>
#include <hip/hip_bf16.h>
#include <math.h>

// ---------------------------------------------------------------------------
// Shapes (compile-time constants from the reference)
// ---------------------------------------------------------------------------
#define BB   256
#define TT   77
#define EE   1280
#define HH   20
#define DD   64
#define NTOK (BB * TT)        // 19712 = 616 * 32
#define N2   (BB * 2 * TT)    // 39424 = 1232 * 32
#define QSCALE 0.125f         // D^-0.5
#define NEGMASK (-1e9f)

typedef __attribute__((ext_vector_type(16))) __bf16 v16bf;
typedef __attribute__((ext_vector_type(8)))  float  v8f;
typedef __attribute__((ext_vector_type(4)))  unsigned v4u;
typedef __attribute__((ext_vector_type(8)))  unsigned v8u;

// ---------------------------------------------------------------------------
// Helpers
// ---------------------------------------------------------------------------
__device__ __forceinline__ float bf2f(__bf16 x) { return (float)x; }
__device__ __forceinline__ __bf16 f2bf(float x) { return (__bf16)x; }

__device__ __forceinline__ float gelu_tanh(float x) {
    const float c0 = 0.7978845608028654f;   // sqrt(2/pi)
    const float c1 = 0.044715f;
    return 0.5f * x * (1.0f + tanhf(c0 * (x + c1 * x * x * x)));
}

// A-matrix fragment (16 x 32 bf16), rows contiguous with `stride` elements.
// CDNA5 layout: lane<16 holds row=lane, K {k0..k0+7, k0+16..k0+23};
//               lane>=16 holds row=lane-16, K {k0+8..k0+15, k0+24..k0+31}.
__device__ __forceinline__ v16bf load_fragA(const __bf16* base, int stride,
                                            int row0, int k0, int lane) {
    const int half = lane >> 4;
    const int r    = lane & 15;
    const __bf16* p = base + (size_t)(row0 + r) * stride + k0 + half * 8;
    union { v16bf v; uint4 u[2]; } f;
    f.u[0] = *(const uint4*)(p);
    f.u[1] = *(const uint4*)(p + 16);
    return f.v;
}

// B-matrix fragment (32 x 16 bf16) where column n of B is stored contiguously
// at base + n*stride (weight-row / transposed layout).
// CDNA5 layout: lane<16 holds col=lane, K {k0..k0+7, k0+16..k0+23};
//               lane>=16 holds col=lane-16, K {k0+8..k0+15, k0+24..k0+31}.
__device__ __forceinline__ v16bf load_fragB(const __bf16* base, int stride,
                                            int col0, int k0, int lane) {
    const int half = lane >> 4;
    const int r    = lane & 15;
    const __bf16* p = base + (size_t)(col0 + r) * stride + k0 + half * 8;
    union { v16bf v; uint4 u[2]; } f;
    f.u[0] = *(const uint4*)(p);
    f.u[1] = *(const uint4*)(p + 16);
    return f.v;
}

// ---------------------------------------------------------------------------
// fp32 -> bf16 cast
// ---------------------------------------------------------------------------
__global__ void cast_f32_to_bf16(const float* __restrict__ in,
                                 __bf16* __restrict__ out, long n) {
    long i = (long)blockIdx.x * blockDim.x + threadIdx.x;
    if (i < n) out[i] = f2bf(in[i]);
}

// ---------------------------------------------------------------------------
// WMMA GEMM with TDM-staged A tile:
//   Out[M,E] = act( A[M,K] @ W[E,K]^T + bias[E] )
// Block = 256 threads (8 waves). Block tile = 32 rows x 256 cols.
// Wave (mi = w>>2, ni = w&3) computes a 16x64 sub-tile: 4 wmma per A-frag.
// A tile (32 x Kdim bf16) is DMA'd into LDS by the Tensor Data Mover.
// Epilogue: 0 = bf16, 1 = bf16 * QSCALE, 2 = bf16 gelu, 3 = f32 out
// ---------------------------------------------------------------------------
template <int EP>
__global__ void gemm_wmma_tdm(const __bf16* __restrict__ A,
                              const __bf16* __restrict__ W,
                              const float* __restrict__ bias,
                              void* __restrict__ Out,
                              int Kdim, int Ncols, int Mtot) {
    __shared__ __align__(16) __bf16 sA[32 * EE];   // 80 KB (Kdim <= EE)

    const int lane   = threadIdx.x & 31;
    const int wave   = threadIdx.x >> 5;
    const int Mbase  = blockIdx.x * 32;
    const int mi     = wave >> 2;                  // 0..1 -> row tile
    const int NbaseW = blockIdx.y * 256 + (wave & 3) * 64;

    // ---- TDM: DMA the 32 x Kdim A tile into LDS (wave 0 issues) ----------
    if (wave == 0) {
        const unsigned lds = (unsigned)(unsigned long long)(&sA[0]);
        const unsigned long long ga =
            (unsigned long long)(const void*)(A + (size_t)Mbase * Kdim);
        v4u g0;
        g0.x = 1u;                                        // count = 1 valid D#
        g0.y = lds;                                       // lds_addr [63:32]
        g0.z = (unsigned)(ga & 0xffffffffull);            // global_addr lo
        g0.w = (unsigned)((ga >> 32) & 0x01ffffffull)     // global_addr hi
               | (2u << 30);                              // type = 2 (image)
        v8u g1;
        g1.s0 = (1u << 16);                               // data_size = 2 B
        g1.s1 = ((unsigned)Kdim & 0xffffu) << 16;         // tensor_dim0 lo16
        g1.s2 = (((unsigned)Kdim >> 16) & 0xffffu)        // tensor_dim0 hi16
                | (((unsigned)Mtot & 0xffffu) << 16);     // tensor_dim1 lo16
        g1.s3 = (((unsigned)Mtot >> 16) & 0xffffu)        // tensor_dim1 hi16
                | (((unsigned)Kdim & 0xffffu) << 16);     // tile_dim0 = Kdim
        g1.s4 = 32u;                                      // tile_dim1 = 32
        g1.s5 = (unsigned)Kdim;                           // dim0_stride lo32
        g1.s6 = 0u;                                       // stride hi / dim1_stride lo
        g1.s7 = 0u;
        asm volatile("tensor_load_to_lds %0, %1" :: "s"(g0), "s"(g1) : "memory");
        __builtin_amdgcn_s_wait_tensorcnt(0);
    }
    __syncthreads();

    // ---- main K loop: one A-frag feeds 4 WMMA (16x64 per wave) -----------
    v8f c0 = {}, c1 = {}, c2 = {}, c3 = {};
    for (int k0 = 0; k0 < Kdim; k0 += 32) {
        if (k0 + 64 < Kdim) {
            __builtin_prefetch(W + (size_t)(NbaseW + (lane & 15)) * Kdim + k0 + 64, 0, 1);
        }
        v16bf a = load_fragA(sA, Kdim, mi * 16, k0, lane);
        v16bf b0 = load_fragB(W, Kdim, NbaseW,      k0, lane);
        c0 = __builtin_amdgcn_wmma_f32_16x16x32_bf16(false, a, false, b0, (short)0, c0, false, false);
        v16bf b1 = load_fragB(W, Kdim, NbaseW + 16, k0, lane);
        c1 = __builtin_amdgcn_wmma_f32_16x16x32_bf16(false, a, false, b1, (short)0, c1, false, false);
        v16bf b2 = load_fragB(W, Kdim, NbaseW + 32, k0, lane);
        c2 = __builtin_amdgcn_wmma_f32_16x16x32_bf16(false, a, false, b2, (short)0, c2, false, false);
        v16bf b3 = load_fragB(W, Kdim, NbaseW + 48, k0, lane);
        c3 = __builtin_amdgcn_wmma_f32_16x16x32_bf16(false, a, false, b3, (short)0, c3, false, false);
    }

    // ---- epilogue ---------------------------------------------------------
    const int half = lane >> 4;
    const int r4   = lane & 15;
    v8f acc[4] = {c0, c1, c2, c3};
#pragma unroll
    for (int j = 0; j < 4; ++j) {
        const int col  = NbaseW + j * 16 + r4;
        const float bc = bias[col];
#pragma unroll
        for (int r = 0; r < 8; ++r) {
            const int row = Mbase + mi * 16 + half * 8 + r;
            float v = acc[j][r] + bc;
            if (EP == 1) v *= QSCALE;
            if (EP == 2) v = gelu_tanh(v);
            if (EP == 3) {
                ((float*)Out)[(size_t)row * Ncols + col] = v;
            } else {
                ((__bf16*)Out)[(size_t)row * Ncols + col] = f2bf(v);
            }
        }
    }
}

// ---------------------------------------------------------------------------
// Attention: one block per (b, h, pass). pass 0 = causal self-attn (writes
// zcat rows 77..153), pass 1 = adapter cross-attn (writes zcat rows 0..76).
// ---------------------------------------------------------------------------
__global__ void attention_wmma(const __bf16* __restrict__ q,
                               const __bf16* __restrict__ k,
                               const __bf16* __restrict__ v,
                               const __bf16* __restrict__ ka,
                               const __bf16* __restrict__ va,
                               __bf16* __restrict__ zcat) {
    __shared__ __align__(16) __bf16 sQ[80 * 64];
    __shared__ __align__(16) __bf16 sK[80 * 64];
    __shared__ __align__(16) __bf16 sVt[64 * 96];   // transposed: [d][s]
    __shared__ float                sL[80 * 80];    // fp32 logits
    __shared__ __align__(16) __bf16 sP[80 * 96];    // bf16 probs, K-padded

    const int tid  = threadIdx.x;
    const int lane = tid & 31;
    const int wave = tid >> 5;
    const int bh   = blockIdx.x;
    const int b    = bh / HH;
    const int h    = bh % HH;
    const int pass = blockIdx.y;          // 0 = self (mask), 1 = adapter

    const __bf16* qp = q + (size_t)(b * TT) * EE + h * DD;
    const __bf16* kp = (pass == 0 ? k : ka) + (size_t)(b * TT) * EE + h * DD;
    const __bf16* vp = (pass == 0 ? v : va) + (size_t)(b * TT) * EE + h * DD;

    for (int idx = tid; idx < 80 * 64; idx += 256) {
        const int s = idx >> 6, d = idx & 63;
        sQ[idx] = (s < TT) ? qp[(size_t)s * EE + d] : f2bf(0.0f);
        sK[idx] = (s < TT) ? kp[(size_t)s * EE + d] : f2bf(0.0f);
    }
    for (int idx = tid; idx < 96 * 64; idx += 256) {
        const int s = idx / 64, d = idx & 63;
        sVt[d * 96 + s] = (s < TT) ? vp[(size_t)s * EE + d] : f2bf(0.0f);
    }
    __syncthreads();

    // ---- logits: 5x5 tiles of 16x16, K = 64 -> two WMMA steps ----
    for (int t = wave; t < 25; t += 8) {
        const int mi = t / 5, ni = t % 5;
        v8f c = {};
#pragma unroll
        for (int k0 = 0; k0 < 64; k0 += 32) {
            v16bf a  = load_fragA(sQ, 64, mi * 16, k0, lane);
            v16bf bb = load_fragB(sK, 64, ni * 16, k0, lane);
            c = __builtin_amdgcn_wmma_f32_16x16x32_bf16(false, a, false, bb,
                                                        (short)0, c, false, false);
        }
        const int half = lane >> 4;
        const int col  = ni * 16 + (lane & 15);
#pragma unroll
        for (int r = 0; r < 8; ++r) {
            sL[(mi * 16 + half * 8 + r) * 80 + col] = c[r];
        }
    }
    __syncthreads();

    // ---- softmax (fp32), one wave per row, causal mask for pass 0 ----
    for (int row = wave; row < 80; row += 8) {
        if (row >= TT) {
            for (int j = lane; j < 96; j += 32) sP[row * 96 + j] = f2bf(0.0f);
            continue;
        }
        float m = -3.4e38f;
        for (int j = lane; j < TT; j += 32) {
            float x = sL[row * 80 + j];
            if (pass == 0 && j > row) x += NEGMASK;
            sL[row * 80 + j] = x;
            m = fmaxf(m, x);
        }
#pragma unroll
        for (int off = 16; off > 0; off >>= 1) m = fmaxf(m, __shfl_xor(m, off));
        float s = 0.0f;
        for (int j = lane; j < TT; j += 32) {
            float e = __expf(sL[row * 80 + j] - m);
            sL[row * 80 + j] = e;
            s += e;
        }
#pragma unroll
        for (int off = 16; off > 0; off >>= 1) s += __shfl_xor(s, off);
        const float inv = 1.0f / s;
        for (int j = lane; j < 96; j += 32) {
            sP[row * 96 + j] = (j < TT) ? f2bf(sL[row * 80 + j] * inv) : f2bf(0.0f);
        }
    }
    __syncthreads();

    // ---- ctx = P(80x96) @ V(96x64): 5x4 tiles, K = 96 -> three WMMA steps --
    __bf16* outbase = zcat + (size_t)b * (2 * TT) * EE
                      + (pass == 0 ? TT : 0) * (size_t)EE + h * DD;
    for (int t = wave; t < 20; t += 8) {
        const int mi = t / 4, nj = t % 4;
        v8f c = {};
#pragma unroll
        for (int k0 = 0; k0 < 96; k0 += 32) {
            v16bf a  = load_fragA(sP, 96, mi * 16, k0, lane);
            v16bf bb = load_fragB(sVt, 96, nj * 16, k0, lane);
            c = __builtin_amdgcn_wmma_f32_16x16x32_bf16(false, a, false, bb,
                                                        (short)0, c, false, false);
        }
        const int half = lane >> 4;
        const int col  = nj * 16 + (lane & 15);
#pragma unroll
        for (int r = 0; r < 8; ++r) {
            const int row = mi * 16 + half * 8 + r;
            if (row < TT) outbase[(size_t)row * EE + col] = f2bf(c[r]);
        }
    }
}

// ---------------------------------------------------------------------------
// Token mixer: Out[b,t,e] = act( sum_s X[b,s,e] * Wt[t,s] + bias[t] ) (+res)
// ---------------------------------------------------------------------------
__global__ void token_mix(const __bf16* __restrict__ X,
                          const float* __restrict__ Wt,
                          const float* __restrict__ bias,
                          const __bf16* __restrict__ residual,
                          __bf16* __restrict__ Out,
                          int S, int Tout,
                          long xBS, long rBS, long oBS, int do_gelu) {
    __shared__ __bf16 Xs[154 * 64];
    const int tid = threadIdx.x;
    const int b   = blockIdx.y;
    const int ec  = blockIdx.x * 64;

    for (int idx = tid; idx < S * 64; idx += 256) {
        const int s = idx / 64, e = idx & 63;
        Xs[idx] = X[(size_t)b * xBS + (size_t)s * EE + ec + e];
    }
    __syncthreads();

    for (int idx = tid; idx < Tout * 64; idx += 256) {
        const int t = idx / 64, e = idx & 63;
        float acc = bias[t];
        for (int s = 0; s < S; ++s) {
            acc += bf2f(Xs[s * 64 + e]) * Wt[t * S + s];
        }
        if (do_gelu) acc = gelu_tanh(acc);
        if (residual) acc += bf2f(residual[(size_t)b * rBS + (size_t)t * EE + ec + e]);
        Out[(size_t)b * oBS + (size_t)t * EE + ec + e] = f2bf(acc);
    }
}

// ---------------------------------------------------------------------------
// Host-side launch
// ---------------------------------------------------------------------------
extern "C" void kernel_launch(void* const* d_in, const int* in_sizes, int n_in,
                              void* d_out, int out_size, void* d_ws, size_t ws_size,
                              hipStream_t stream) {
    (void)in_sizes; (void)n_in; (void)out_size; (void)ws_size;

    const float* hidden = (const float*)d_in[0];
    const float* embeds = (const float*)d_in[1];
    // d_in[2] = causal_mask (reconstructed analytically)
    const float* q_w  = (const float*)d_in[3];  const float* q_b  = (const float*)d_in[4];
    const float* k_w  = (const float*)d_in[5];  const float* k_b  = (const float*)d_in[6];
    const float* v_w  = (const float*)d_in[7];  const float* v_b  = (const float*)d_in[8];
    const float* o_w  = (const float*)d_in[9];  const float* o_b  = (const float*)d_in[10];
    const float* ka_w = (const float*)d_in[11]; const float* ka_b = (const float*)d_in[12];
    const float* va_w = (const float*)d_in[13]; const float* va_b = (const float*)d_in[14];
    const float* zc1_w = (const float*)d_in[15]; const float* zc1_b = (const float*)d_in[16];
    const float* zt1_w = (const float*)d_in[17]; const float* zt1_b = (const float*)d_in[18];
    const float* zc2_w = (const float*)d_in[19]; const float* zc2_b = (const float*)d_in[20];
    const float* zt2_w = (const float*)d_in[21]; const float* zt2_b = (const float*)d_in[22];

    const size_t NE = (size_t)NTOK * EE;     // 25,231,360 elems
    const size_t WE = (size_t)EE * EE;       //  1,638,400 elems

    __bf16* ws = (__bf16*)d_ws;
    __bf16* hbf   = ws;                      // hidden (N,E)
    __bf16* ebf   = hbf + NE;                // embeds (N,E)
    __bf16* wq    = ebf + NE;
    __bf16* wk    = wq  + WE;
    __bf16* wv    = wk  + WE;
    __bf16* wka   = wv  + WE;
    __bf16* wva   = wka + WE;
    __bf16* wo    = wva + WE;
    __bf16* wc1   = wo  + WE;
    __bf16* wc2   = wc1 + WE;
    __bf16* pool  = wc2 + WE;                // 5*NE elems, reused
    __bf16* qbf   = pool;                    // q  (N,E)
    __bf16* kbf   = pool + NE;               // k
    __bf16* vbf   = pool + 2 * NE;           // v
    __bf16* kabf  = pool + 3 * NE;           // ka
    __bf16* vabf  = pool + 4 * NE;           // va
    __bf16* zcat  = pool + 5 * NE;           // (B, 2T, E) = 2*NE elems
    // after attention q..va are dead -> reuse pool for zipper intermediates
    __bf16* x1 = pool;                       // gelu(zcat@zc1^T)  (B,2T,E) 2*NE
    __bf16* x2 = pool + 2 * NE;              // token mix 1       (B,T,E)
    __bf16* x3 = pool + 3 * NE;              // zc2 out           (B,T,E)
    __bf16* yb = pool + 4 * NE;              // x4 + self_ctx     (B,T,E)

    dim3 blk(256);
    auto cast = [&](const float* src, __bf16* dst, long n) {
        cast_f32_to_bf16<<<dim3((unsigned)((n + 255) / 256)), blk, 0, stream>>>(src, dst, n);
    };
    cast(hidden, hbf, (long)NE);
    cast(embeds, ebf, (long)NE);
    cast(q_w,  wq,  (long)WE);  cast(k_w,  wk,  (long)WE);
    cast(v_w,  wv,  (long)WE);  cast(ka_w, wka, (long)WE);
    cast(va_w, wva, (long)WE);  cast(o_w,  wo,  (long)WE);
    cast(zc1_w, wc1, (long)WE); cast(zc2_w, wc2, (long)WE);

    const dim3 gN(NTOK / 32, EE / 256);      // (616, 5)
    const dim3 g2(N2 / 32,  EE / 256);       // (1232, 5)

    // projections
    gemm_wmma_tdm<1><<<gN, blk, 0, stream>>>(hbf, wq,  q_b,  qbf,  EE, EE, NTOK);
    gemm_wmma_tdm<0><<<gN, blk, 0, stream>>>(hbf, wk,  k_b,  kbf,  EE, EE, NTOK);
    gemm_wmma_tdm<0><<<gN, blk, 0, stream>>>(hbf, wv,  v_b,  vbf,  EE, EE, NTOK);
    gemm_wmma_tdm<0><<<gN, blk, 0, stream>>>(ebf, wka, ka_b, kabf, EE, EE, NTOK);
    gemm_wmma_tdm<0><<<gN, blk, 0, stream>>>(ebf, wva, va_b, vabf, EE, EE, NTOK);

    // attention (self + adapter) -> zcat = [adapter | self] along tokens
    attention_wmma<<<dim3(BB * HH, 2), blk, 0, stream>>>(qbf, kbf, vbf, kabf, vabf, zcat);

    // zipper
    gemm_wmma_tdm<2><<<g2, blk, 0, stream>>>(zcat, wc1, zc1_b, x1, EE, EE, N2);
    token_mix<<<dim3(EE / 64, BB), blk, 0, stream>>>(x1, zt1_w, zt1_b, nullptr, x2,
                                                     2 * TT, TT,
                                                     (long)2 * TT * EE, 0L,
                                                     (long)TT * EE, 1);
    gemm_wmma_tdm<0><<<gN, blk, 0, stream>>>(x2, wc2, zc2_b, x3, EE, EE, NTOK);
    token_mix<<<dim3(EE / 64, BB), blk, 0, stream>>>(x3, zt2_w, zt2_b,
                                                     zcat + (size_t)TT * EE, yb,
                                                     TT, TT,
                                                     (long)TT * EE, (long)2 * TT * EE,
                                                     (long)TT * EE, 0);

    // output projection -> fp32 d_out
    gemm_wmma_tdm<3><<<gN, blk, 0, stream>>>(yb, wo, o_b, d_out, EE, EE, NTOK);
}
